// TransformerBlock_9603546873923
// MI455X (gfx1250) — matrix operand order repaired and checked by reading, not compiled, not run on hardware
//
#include <hip/hip_runtime.h>
#include <hip/hip_bf16.h>
#include <cstdint>
#include <cstddef>

typedef __bf16 bf16;
typedef __bf16 v16bf __attribute__((ext_vector_type(16)));
typedef __bf16 v8bf  __attribute__((ext_vector_type(8)));
typedef float  v8f   __attribute__((ext_vector_type(8)));

#define DEV __device__ __forceinline__

#ifndef USE_ASYNC_LDS
#define USE_ASYNC_LDS 1
#endif

static constexpr int D_MODEL = 1024;
static constexpr int N_HEADS = 16;
static constexpr int D_HEAD  = 64;
static constexpr int D_FF    = 4096;
static constexpr int SEQ     = 2048;
static constexpr int BATCH   = 2;
static constexpr int ROWS    = BATCH * SEQ; // 4096

DEV v8f zero8() {
  v8f z;
#pragma unroll
  for (int i = 0; i < 8; ++i) z[i] = 0.0f;
  return z;
}

union FragBF { v16bf v; v8bf h[2]; };

// A-fragment (16x32 bf16, M x K): lane<16 -> row=lane, K = 0..7 & 16..23;
// lane>=16 -> row=lane-16, K = 8..15 & 24..31.  base_row points at [row][k0].
DEV v16bf load_frag_a(const bf16* base_row, int kbA) {
  FragBF u;
  u.h[0] = *reinterpret_cast<const v8bf*>(base_row + kbA);
  u.h[1] = *reinterpret_cast<const v8bf*>(base_row + kbA + 16);
  return u.v;
}
// B-fragment (32x16 bf16, K x N) sourced from row-major [N,K] storage:
// lane<16 -> col=lane, K = 0..7 & 16..23; lane>=16 -> col=lane-16, K = 8..15 & 24..31.
DEV v16bf load_frag_b(const bf16* base_col, int kbB) {
  FragBF u;
  u.h[0] = *reinterpret_cast<const v8bf*>(base_col + kbB);
  u.h[1] = *reinterpret_cast<const v8bf*>(base_col + kbB + 16);
  return u.v;
}

DEV v8f wmma_bf16(v16bf a, v16bf b, v8f c) {
  return __builtin_amdgcn_wmma_f32_16x16x32_bf16(false, a, false, b, (short)0, c,
                                                 false, false);
}

// ---------------------------------------------------------------- fp32 -> bf16
__global__ __launch_bounds__(256)
void cvt_bf16_kernel(const float* __restrict__ in, bf16* __restrict__ out, int n) {
  int i = blockIdx.x * 256 + threadIdx.x;
  const int stride = gridDim.x * 256;
  for (; i < n; i += stride) out[i] = (bf16)in[i];
}

// -------------------------------------------------------- generic WMMA GEMM
// C[M,N] = A[M,K] @ W[N,K]^T + bias.
// Wave tile 64x64 (16 WMMA per K=32 step), block = 4 waves stacked on M
// (block tile 256x64).  The 64x32 B tile is shared by all 4 waves and staged
// into LDS via gfx1250 global_load_async_to_lds_b128 (double buffered).
// STORE_VT: scatter output as bf16 Vt[B,H,dk,S] (keys contiguous) for PV matmul.
template<bool OUT_BF16, bool RELU, bool STORE_VT>
__global__ __launch_bounds__(128)
void gemm_bf16_kernel(const bf16* __restrict__ A, int lda,
                      const bf16* __restrict__ W, int ldw,
                      const float* __restrict__ bias,
                      void* __restrict__ outv, int ldc, int K) {
  __shared__ __align__(16) bf16 bt[2][64 * 32];  // 2 x 4 KB B tiles
  const int tid    = threadIdx.x;
  const int lane   = tid & 31;
  const int wave   = tid >> 5;
  const int half16 = lane >> 4;
  const int l15    = lane & 15;
  const int tm0 = blockIdx.x * 256 + wave * 64;
  const int tn0 = blockIdx.y * 64;
  const int kbA = half16 ? 8 : 0;
  const int kbB = half16 ? 8 : 0;

  v8f acc[4][4];
#pragma unroll
  for (int m = 0; m < 4; ++m)
#pragma unroll
    for (int n = 0; n < 4; ++n) acc[m][n] = zero8();

  const bf16* arow[4];
#pragma unroll
  for (int m = 0; m < 4; ++m)
    arow[m] = A + (size_t)(tm0 + m * 16 + l15) * lda;

  // Cooperative staging of the 64x32 bf16 B tile (4 KB = 256 16-byte chunks,
  // 2 chunks per thread).  LDS layout mirrors global: [col][k0..k0+31].
  auto stage = [&](int buf, int k0) {
#pragma unroll
    for (int i = 0; i < 2; ++i) {
      const int c    = tid + i * 128;  // chunk id 0..255
      const int col  = c >> 2;
      const int part = c & 3;          // 16B sub-chunk (8 bf16)
      const bf16* src = W + (size_t)(tn0 + col) * ldw + k0 + part * 8;
      bf16* dst = &bt[buf][col * 32 + part * 8];
#if defined(__gfx1250__) && USE_ASYNC_LDS
      // LDS byte offset of dst within the workgroup allocation (single
      // __shared__ object -> allocation starts at offset 0).
      const unsigned ldsoff =
          (unsigned)((const char*)dst - (const char*)&bt[0][0]);
      asm volatile("global_load_async_to_lds_b128 %0, %1, off"
                   :: "v"(ldsoff), "v"((unsigned long long)(uintptr_t)src)
                   : "memory");
#else
      *reinterpret_cast<v8bf*>(dst) = *reinterpret_cast<const v8bf*>(src);
#endif
    }
  };

  stage(0, 0);
#if defined(__gfx1250__) && USE_ASYNC_LDS
  asm volatile("s_wait_asynccnt 0x0" ::: "memory");
#endif
  __syncthreads();

  int buf = 0;
  for (int k0 = 0; k0 < K; k0 += 32) {
    if (k0 + 32 < K) {
      stage(buf ^ 1, k0 + 32);                     // async fill of next tile
      __builtin_prefetch(arow[0] + k0 + 32, 0, 1); // global_prefetch_b8 hint
    }

    v16bf af[4];
#pragma unroll
    for (int m = 0; m < 4; ++m) af[m] = load_frag_a(arow[m] + k0, kbA);

    const bf16* bbase = &bt[buf][0];
#pragma unroll
    for (int n = 0; n < 4; ++n) {
      // B fragment from LDS (ds_load_b128 pairs)
      FragBF u;
      const bf16* bc = bbase + (n * 16 + l15) * 32 + kbB;
      u.h[0] = *reinterpret_cast<const v8bf*>(bc);
      u.h[1] = *reinterpret_cast<const v8bf*>(bc + 16);
      const v16bf bfr = u.v;
#pragma unroll
      for (int m = 0; m < 4; ++m) acc[m][n] = wmma_bf16(af[m], bfr, acc[m][n]);
    }

#if defined(__gfx1250__) && USE_ASYNC_LDS
    asm volatile("s_wait_asynccnt 0x0" ::: "memory");
#endif
    __syncthreads();
    buf ^= 1;
  }

  // C/D layout: vgpr r -> row r (lanes 0-15) / row r+8 (lanes 16-31), col = l15.
#pragma unroll
  for (int n = 0; n < 4; ++n) {
    const int col = tn0 + n * 16 + l15;
    const float bb = bias[col];
#pragma unroll
    for (int m = 0; m < 4; ++m) {
#pragma unroll
      for (int r = 0; r < 8; ++r) {
        const int row = tm0 + m * 16 + r + half16 * 8;
        float v = acc[m][n][r] + bb;
        if (RELU) v = fmaxf(v, 0.0f);
        if (STORE_VT) {
          const int b = row >> 11, s = row & (SEQ - 1);
          const int h = col >> 6, dc = col & (D_HEAD - 1);
          ((bf16*)outv)[(((size_t)(b * N_HEADS + h) * D_HEAD + dc) << 11) + s] =
              (bf16)v;
        } else if (OUT_BF16) {
          ((bf16*)outv)[(size_t)row * ldc + col] = (bf16)v;
        } else {
          ((float*)outv)[(size_t)row * ldc + col] = v;
        }
      }
    }
  }
}

// ------------------------------------------------- flash attention (per head)
// One wave owns a 16-row Q tile; loops over 32 keys per iteration.
__global__ __launch_bounds__(128)
void attention_kernel(const bf16* __restrict__ Q, const bf16* __restrict__ Km,
                      const bf16* __restrict__ Vt, bf16* __restrict__ ctx) {
  __shared__ __align__(16) bf16 plds[4][16 * 32];
  const int lane   = threadIdx.x & 31;
  const int wave   = threadIdx.x >> 5;
  const int half16 = lane >> 4;
  const int l15    = lane & 15;
  const int gw = blockIdx.x * 4 + wave;       // 0..4095
  const int qt = gw & 127;                    // S/16 tiles
  const int bh = gw >> 7;
  const int b  = bh >> 4;
  const int h  = bh & 15;
  const size_t rowBase = (size_t)b * SEQ;
  const int kbA = half16 ? 8 : 0;
  const int kbB = half16 ? 8 : 0;

  // Q fragments: 16 rows x dk=64 -> two K=32 chunks, resident for whole loop
  v16bf qf[2];
  {
    const bf16* qrow = Q + (rowBase + qt * 16 + l15) * D_MODEL + h * D_HEAD;
    qf[0] = load_frag_a(qrow, kbA);
    qf[1] = load_frag_a(qrow + 32, kbA);
  }

  float mrow[8], lrow[8];
  v8f oacc[4];
#pragma unroll
  for (int r = 0; r < 8; ++r) { mrow[r] = -3.0e38f; lrow[r] = 0.0f; }
#pragma unroll
  for (int c = 0; c < 4; ++c) oacc[c] = zero8();

  bf16* pw = plds[wave];

  for (int kb0 = 0; kb0 < SEQ; kb0 += 32) {
    // scores: 16 q-rows x 32 keys, contraction over dk=64
    v8f st0 = zero8(), st1 = zero8();
    {
      const bf16* kc0 = Km + (rowBase + kb0 + l15) * D_MODEL + h * D_HEAD;
      const bf16* kc1 = kc0 + 16 * D_MODEL;
      st0 = wmma_bf16(qf[0], load_frag_b(kc0, kbB), st0);
      st0 = wmma_bf16(qf[1], load_frag_b(kc0 + 32, kbB), st0);
      st1 = wmma_bf16(qf[0], load_frag_b(kc1, kbB), st1);
      st1 = wmma_bf16(qf[1], load_frag_b(kc1 + 32, kbB), st1);
    }
    st0 = st0 * 0.125f;  // 1/sqrt(64)
    st1 = st1 * 0.125f;

    // online softmax, per C-layout row register
#pragma unroll
    for (int r = 0; r < 8; ++r) {
      float mx = fmaxf(st0[r], st1[r]);
#pragma unroll
      for (int sm = 8; sm >= 1; sm >>= 1) mx = fmaxf(mx, __shfl_xor(mx, sm, 32));
      const float mnew  = fmaxf(mrow[r], mx);
      const float alpha = __expf(mrow[r] - mnew);
      const float p0 = __expf(st0[r] - mnew);
      const float p1 = __expf(st1[r] - mnew);
      float rs = p0 + p1;
#pragma unroll
      for (int sm = 8; sm >= 1; sm >>= 1) rs += __shfl_xor(rs, sm, 32);
      lrow[r] = lrow[r] * alpha + rs;
      mrow[r] = mnew;
#pragma unroll
      for (int c = 0; c < 4; ++c) oacc[c][r] *= alpha;
      const int prow = r + half16 * 8;
      pw[prow * 32 + l15]      = (bf16)p0;   // ds_store_b16
      pw[prow * 32 + l15 + 16] = (bf16)p1;
    }
    __builtin_amdgcn_wave_barrier();
#if defined(__gfx1250__)
    asm volatile("s_wait_dscnt 0x0" ::: "memory");  // LDS writes visible wave-wide
#endif

    // P (16x32) reread from LDS in A-fragment layout
    v16bf pf;
    {
      FragBF u;
      u.h[0] = *reinterpret_cast<const v8bf*>(pw + l15 * 32 + kbA);
      u.h[1] = *reinterpret_cast<const v8bf*>(pw + l15 * 32 + kbA + 16);
      pf = u.v;
    }
    __builtin_amdgcn_wave_barrier();

    // O += P @ V : Vt is [B,H,dk,S] so keys are contiguous per dk column
#pragma unroll
    for (int c = 0; c < 4; ++c) {
      const bf16* vc =
          Vt + ((size_t)bh * D_HEAD + c * 16 + l15) * SEQ + kb0;
      oacc[c] = wmma_bf16(pf, load_frag_b(vc, kbB), oacc[c]);
    }
  }

  // normalize and store context (bf16, [B*S, D] layout)
#pragma unroll
  for (int r = 0; r < 8; ++r) {
    const float inv = 1.0f / lrow[r];
#pragma unroll
    for (int c = 0; c < 4; ++c) oacc[c][r] *= inv;
  }
  const size_t orow0 = rowBase + (size_t)qt * 16;
#pragma unroll
  for (int c = 0; c < 4; ++c) {
    const int col = h * D_HEAD + c * 16 + l15;
#pragma unroll
    for (int r = 0; r < 8; ++r) {
      const size_t row = orow0 + r + half16 * 8;
      ctx[row * D_MODEL + col] = (bf16)oacc[c][r];
    }
  }
}

// ------------------------------------------- residual add + LayerNorm (fused)
template<bool MAKE_BF16>
__global__ __launch_bounds__(256)
void add_ln_kernel(const float* __restrict__ a, const float* __restrict__ b,
                   const float* __restrict__ g, const float* __restrict__ be,
                   float* __restrict__ y, bf16* __restrict__ yb) {
  const int row = blockIdx.x;
  const int tid = threadIdx.x;
  const size_t base = (size_t)row * D_MODEL;
  float v[4];
  float s = 0.0f, q = 0.0f;
#pragma unroll
  for (int i = 0; i < 4; ++i) {
    const int c = tid + i * 256;
    const float t = a[base + c] + b[base + c];
    v[i] = t; s += t; q += t * t;
  }
#pragma unroll
  for (int sm = 16; sm >= 1; sm >>= 1) {
    s += __shfl_xor(s, sm, 32);
    q += __shfl_xor(q, sm, 32);
  }
  __shared__ float s1[8], s2[8];
  const int wid = tid >> 5;
  if ((tid & 31) == 0) { s1[wid] = s; s2[wid] = q; }
  __syncthreads();
  float ts = 0.0f, tq = 0.0f;
#pragma unroll
  for (int w = 0; w < 8; ++w) { ts += s1[w]; tq += s2[w]; }
  const float mu   = ts * (1.0f / (float)D_MODEL);
  const float var  = tq * (1.0f / (float)D_MODEL) - mu * mu;
  const float rstd = rsqrtf(var + 1e-5f);
#pragma unroll
  for (int i = 0; i < 4; ++i) {
    const int c = tid + i * 256;
    const float o = (v[i] - mu) * rstd * g[c] + be[c];
    y[base + c] = o;
    if (MAKE_BF16) yb[base + c] = (bf16)o;
  }
}

// ---------------------------------------------------------------------- host
extern "C" void kernel_launch(void* const* d_in, const int* in_sizes, int n_in,
                              void* d_out, int out_size, void* d_ws,
                              size_t ws_size, hipStream_t stream) {
  (void)in_sizes; (void)n_in; (void)out_size; (void)ws_size;
  const float* x   = (const float*)d_in[0];
  const float* Wq  = (const float*)d_in[1];
  const float* bq  = (const float*)d_in[2];
  const float* Wk  = (const float*)d_in[3];
  const float* bk  = (const float*)d_in[4];
  const float* Wv  = (const float*)d_in[5];
  const float* bv  = (const float*)d_in[6];
  const float* Wo  = (const float*)d_in[7];
  const float* bo  = (const float*)d_in[8];
  const float* W1  = (const float*)d_in[9];
  const float* b1  = (const float*)d_in[10];
  const float* W2  = (const float*)d_in[11];
  const float* b2  = (const float*)d_in[12];
  const float* g1  = (const float*)d_in[13];
  const float* be1 = (const float*)d_in[14];
  const float* g2  = (const float*)d_in[15];
  const float* be2 = (const float*)d_in[16];
  float* out = (float*)d_out;

  char* ws = (char*)d_ws;
  const size_t MB = 1ull << 20;
  bf16*  xb  = (bf16*) (ws + 0);        //  8 MB  x in bf16
  bf16*  wqb = (bf16*) (ws + 8 * MB);   //  2 MB
  bf16*  wkb = (bf16*) (ws + 10 * MB);  //  2 MB
  bf16*  wvb = (bf16*) (ws + 12 * MB);  //  2 MB
  bf16*  wob = (bf16*) (ws + 14 * MB);  //  2 MB
  bf16*  w1b = (bf16*) (ws + 16 * MB);  //  8 MB
  bf16*  w2b = (bf16*) (ws + 24 * MB);  //  8 MB
  bf16*  Qb  = (bf16*) (ws + 32 * MB);  //  8 MB
  bf16*  Kb  = (bf16*) (ws + 40 * MB);  //  8 MB
  bf16*  Vt  = (bf16*) (ws + 48 * MB);  //  8 MB  [B,H,dk,S]
  bf16*  ctx = (bf16*) (ws + 56 * MB);  //  8 MB
  float* t32 = (float*)(ws + 64 * MB);  // 16 MB  attn_out, then ff2
  float* x1  = (float*)(ws + 80 * MB);  // 16 MB
  bf16*  x1b = (bf16*) (ws + 96 * MB);  //  8 MB
  bf16*  ff1 = (bf16*) (ws + 104 * MB); // 32 MB

  auto cvt = [&](const float* src, bf16* dst, int n) {
    cvt_bf16_kernel<<<1024, 256, 0, stream>>>(src, dst, n);
  };
  cvt(x,  xb,  ROWS * D_MODEL);
  cvt(Wq, wqb, D_MODEL * D_MODEL);
  cvt(Wk, wkb, D_MODEL * D_MODEL);
  cvt(Wv, wvb, D_MODEL * D_MODEL);
  cvt(Wo, wob, D_MODEL * D_MODEL);
  cvt(W1, w1b, D_FF * D_MODEL);
  cvt(W2, w2b, D_MODEL * D_FF);

  const dim3 blk(128);
  const dim3 gD(ROWS / 256, D_MODEL / 64);
  const dim3 gF(ROWS / 256, D_FF / 64);

  // QKV projections
  gemm_bf16_kernel<true,  false, false><<<gD, blk, 0, stream>>>(
      xb, D_MODEL, wqb, D_MODEL, bq, Qb, D_MODEL, D_MODEL);
  gemm_bf16_kernel<true,  false, false><<<gD, blk, 0, stream>>>(
      xb, D_MODEL, wkb, D_MODEL, bk, Kb, D_MODEL, D_MODEL);
  gemm_bf16_kernel<true,  false, true ><<<gD, blk, 0, stream>>>(
      xb, D_MODEL, wvb, D_MODEL, bv, Vt, 0, D_MODEL);

  // fused flash attention
  attention_kernel<<<1024, 128, 0, stream>>>(Qb, Kb, Vt, ctx);

  // output projection + LN1
  gemm_bf16_kernel<false, false, false><<<gD, blk, 0, stream>>>(
      ctx, D_MODEL, wob, D_MODEL, bo, t32, D_MODEL, D_MODEL);
  add_ln_kernel<true><<<ROWS, 256, 0, stream>>>(x, t32, g1, be1, x1, x1b);

  // FFN + LN2
  gemm_bf16_kernel<true,  true,  false><<<gF, blk, 0, stream>>>(
      x1b, D_MODEL, w1b, D_MODEL, b1, ff1, D_FF, D_MODEL);
  gemm_bf16_kernel<false, false, false><<<gD, blk, 0, stream>>>(
      ff1, D_FF, w2b, D_FF, b2, t32, D_MODEL, D_FF);
  add_ln_kernel<false><<<ROWS, 256, 0, stream>>>(x1, t32, g2, be2, out, nullptr);
}
